// Community2Emb_1219770712879
// MI455X (gfx1250) — matrix operand order repaired and checked by reading, not compiled, not run on hardware
//
#include <hip/hip_runtime.h>

typedef __bf16 v16bf __attribute__((ext_vector_type(16)));
typedef float  v8f   __attribute__((ext_vector_type(8)));

#define GDIM 128   // embedding dimension D
#define GK   32    // number of mixture components K

// ============================================================================
// Stage A: per-component Cholesky, logdet, B = (L^{-1})^T, t = mu*B,
// and packing of B into WMMA B-fragment layout as bf16 hi/lo split.
// One block per component k, 128 threads.
// ============================================================================
__global__ void __launch_bounds__(128)
stageA_kernel(const float* __restrict__ cov, const float* __restrict__ centroid,
              __bf16* __restrict__ Mh, __bf16* __restrict__ Ml,
              float* __restrict__ tvec, float* __restrict__ logdet)
{
    __shared__ float sA[GDIM][GDIM + 1];   // 128 x 129 f32 = 66,048 B (LDS is 320KB/WGP)
    const int k   = blockIdx.x;
    const int tid = threadIdx.x;

    const float* C = cov + (size_t)k * GDIM * GDIM;
    for (int idx = tid; idx < GDIM * GDIM; idx += 128)
        sA[idx >> 7][idx & 127] = C[idx];
    __syncthreads();

    // In-place right-looking Cholesky (lower triangle): cov = L * L^T
    for (int j = 0; j < GDIM; ++j) {
        if (tid == 0) sA[j][j] = sqrtf(sA[j][j]);
        __syncthreads();
        const float invd = 1.0f / sA[j][j];
        const int i = j + 1 + tid;
        if (i < GDIM) sA[i][j] *= invd;
        __syncthreads();
        if (tid > j) {
            const float lij = sA[tid][j];
            for (int c2 = j + 1; c2 <= tid; ++c2)
                sA[tid][c2] -= lij * sA[c2][j];
        }
        __syncthreads();
    }

    if (tid == 0) {
        float s = 0.0f;
        for (int j = 0; j < GDIM; ++j) s += logf(sA[j][j]);
        logdet[k] = 2.0f * s;
    }

    // U = L^{-1} (lower triangular). Thread tid solves column c = tid by
    // forward substitution; U[i][c] (i>c) is stored in the unused upper
    // triangle at sA[c][i]. No cross-thread hazards (disjoint rows/reads).
    {
        const int c = tid;
        const float udiag = 1.0f / sA[c][c];
        for (int i = c + 1; i < GDIM; ++i) {
            float s = sA[i][c] * udiag;
            for (int m = c + 1; m < i; ++m) s += sA[i][m] * sA[c][m];
            sA[c][i] = -s / sA[i][i];
        }
    }
    __syncthreads();

    // B[d][e] = U[e][d] = (e<d) ? 0 : (e==d ? 1/L[d][d] : sA[d][e])
    // t[e] = sum_d mu[d] * B[d][e]   (only d<=e contribute; B upper-tri)
    {
        const int e = tid;
        const float* mu = centroid + (size_t)k * GDIM;
        float s = mu[e] / sA[e][e];
        for (int d = 0; d < e; ++d) s += mu[d] * sA[d][e];
        tvec[k * GDIM + e] = s;
    }

    // Pack B into WMMA B-fragment layout, bf16 hi/lo:
    //   off = (((k*8 + j)*4 + kc)*32 + lane)*16 + h
    //   d   = kc*32 + (lane>>4)*16 + h        (K index within B 32x16 frag)
    //   e   = j*16 + (lane&15)                (N index)
    for (int idx = tid; idx < 8 * 4 * 32 * 16; idx += 128) {
        const int h    = idx & 15;
        const int lane = (idx >> 4) & 31;
        const int kc   = (idx >> 9) & 3;
        const int j    = (idx >> 11) & 7;
        const int d    = kc * 32 + ((lane >> 4) << 4) + h;
        const int e    = j * 16 + (lane & 15);
        float v;
        if (e < d)       v = 0.0f;
        else if (e == d) v = 1.0f / sA[d][d];
        else             v = sA[d][e];
        const __bf16 hi = (__bf16)v;
        const __bf16 lo = (__bf16)(v - (float)hi);
        const size_t off = (size_t)k * 16384 + (size_t)idx;
        Mh[off] = hi;
        Ml[off] = lo;
    }
}

// ============================================================================
// Main kernel: Z = X * B_k via bf16x3 split WMMA, maha = sum_e (Z - t)^2.
// Each wave processes TWO 16-row strips against each B fragment (B reuse in
// registers -> half the B-load traffic per WMMA; loop becomes compute-bound).
// Block = 128 threads (4 waves) -> 128 rows per block. Grid = (ceil(N/128), K).
// ============================================================================
__global__ void __launch_bounds__(128)
maha_kernel(const float* __restrict__ X, const __bf16* __restrict__ Mh,
            const __bf16* __restrict__ Ml, const float* __restrict__ pivec,
            const float* __restrict__ tvec, const float* __restrict__ logdet,
            float* __restrict__ partials, int N)
{
    const int bx   = blockIdx.x;
    const int k    = blockIdx.y;
    const int tid  = threadIdx.x;
    const int wave = tid >> 5;
    const int lane = tid & 31;
    const int n16  = lane & 15;   // row within tile for A-frag; col for C/D
    const int g    = lane >> 4;   // K-half (A-frag) / row group (C/D)

    const int rowBase = bx * 128 + wave * 32;     // this wave: 32 rows
    int r0 = rowBase + n16;                        // strip 0 A-row
    int r1 = rowBase + 16 + n16;                   // strip 1 A-row
    if (r0 >= N) r0 = N - 1;                       // branchless clamps
    if (r1 >= N) r1 = N - 1;                       // (EXEC stays all-ones)

    v8f acc[2][8] = {};                            // 2 strips x 8 col-tiles

    #pragma unroll
    for (int kc = 0; kc < 4; ++kc) {
        // A-fragments: lane holds row r, K = kc*32 + g*16 + h
        v16bf ah[2], al[2];
        #pragma unroll
        for (int s = 0; s < 2; ++s) {
            const int r = (s == 0) ? r0 : r1;
            const float* xrow = X + (size_t)r * GDIM + kc * 32 + g * 16;
            const float4 a0 = *(const float4*)(xrow);
            const float4 a1 = *(const float4*)(xrow + 4);
            const float4 a2 = *(const float4*)(xrow + 8);
            const float4 a3 = *(const float4*)(xrow + 12);
            float xs[16] = { a0.x, a0.y, a0.z, a0.w,  a1.x, a1.y, a1.z, a1.w,
                             a2.x, a2.y, a2.z, a2.w,  a3.x, a3.y, a3.z, a3.w };
            #pragma unroll
            for (int h = 0; h < 16; ++h) {
                const float  x  = xs[h];
                const __bf16 hi = (__bf16)x;
                ah[s][h] = hi;
                al[s][h] = (__bf16)(x - (float)hi);
            }
        }

        #pragma unroll
        for (int j = 0; j < 8; ++j) {
            const size_t off = (size_t)k * 16384 + (size_t)j * 2048
                             + (size_t)kc * 512 + (size_t)lane * 16;
            const v16bf bh = *(const v16bf*)(Mh + off);
            const v16bf bl = *(const v16bf*)(Ml + off);
            // bf16x3 split product, both row strips share bh/bl:
            #pragma unroll
            for (int s = 0; s < 2; ++s) {
                acc[s][j] = __builtin_amdgcn_wmma_f32_16x16x32_bf16(
                    false, ah[s], false, bh, (short)0, acc[s][j], false, false);
                acc[s][j] = __builtin_amdgcn_wmma_f32_16x16x32_bf16(
                    false, ah[s], false, bl, (short)0, acc[s][j], false, false);
                acc[s][j] = __builtin_amdgcn_wmma_f32_16x16x32_bf16(
                    false, al[s], false, bh, (short)0, acc[s][j], false, false);
            }
        }
    }

    // Epilogue: D-tile element (m, n) lives at VGPR v = m%8, lane = n + 16*(m/8).
    float tval[8];
    #pragma unroll
    for (int j = 0; j < 8; ++j) tval[j] = tvec[k * GDIM + j * 16 + n16];

    float local = 0.0f;
    const float constk = (float)GDIM * 1.8378770664093453f + logdet[k];

    #pragma unroll
    for (int s = 0; s < 2; ++s) {
        float srow[8];
        #pragma unroll
        for (int v = 0; v < 8; ++v) {
            float sum = 0.0f;
            #pragma unroll
            for (int j = 0; j < 8; ++j) {
                const float z = acc[s][j][v] - tval[j];
                sum += z * z;
            }
            srow[v] = sum;
        }
        // Reduce across the 16 lanes of a row group (xor masks < 16 keep bit4).
        #pragma unroll
        for (int v = 0; v < 8; ++v) {
            srow[v] += __shfl_xor(srow[v], 1);
            srow[v] += __shfl_xor(srow[v], 2);
            srow[v] += __shfl_xor(srow[v], 4);
            srow[v] += __shfl_xor(srow[v], 8);
        }
        if (n16 == 0) {
            #pragma unroll
            for (int v = 0; v < 8; ++v) {
                const int row = rowBase + s * 16 + g * 8 + v;
                if (row < N) {
                    const float p = pivec[(size_t)row * GK + k];
                    local += p * (-0.5f) * (constk + srow[v]);
                }
            }
        }
    }

    __shared__ float red[128];
    red[tid] = local;
    __syncthreads();
    for (int st = 64; st > 0; st >>= 1) {
        if (tid < st) red[tid] += red[tid + st];
        __syncthreads();
    }
    if (tid == 0) partials[(size_t)k * gridDim.x + bx] = red[0];
}

// ============================================================================
// Deterministic final reduction + scaling by (-BETA/K).
// ============================================================================
__global__ void finalize_kernel(const float* __restrict__ partials, int count,
                                float* __restrict__ out)
{
    __shared__ float red[256];
    float s = 0.0f;
    for (int i = threadIdx.x; i < count; i += 256) s += partials[i];
    red[threadIdx.x] = s;
    __syncthreads();
    for (int st = 128; st > 0; st >>= 1) {
        if ((int)threadIdx.x < st) red[threadIdx.x] += red[threadIdx.x + st];
        __syncthreads();
    }
    if (threadIdx.x == 0) out[0] = red[0] * (-1.0f / (float)GK);
}

// ============================================================================
// Launch
// ============================================================================
extern "C" void kernel_launch(void* const* d_in, const int* in_sizes, int n_in,
                              void* d_out, int out_size, void* d_ws, size_t ws_size,
                              hipStream_t stream)
{
    (void)n_in; (void)out_size; (void)ws_size;
    const float* X    = (const float*)d_in[0];   // node_emb [N,128]
    const float* mu   = (const float*)d_in[1];   // centroid [32,128]
    const float* cov  = (const float*)d_in[2];   // covariance [32,128,128]
    const float* pvec = (const float*)d_in[3];   // pi [N,32]
    const int N = in_sizes[0] / GDIM;

    char* w = (char*)d_ws;
    __bf16* Mh     = (__bf16*)(w);               // 1 MB: B-frags, hi part
    __bf16* Ml     = (__bf16*)(w + (1u << 20));  // 1 MB: B-frags, lo part
    float*  tvec   = (float*)(w + (2u << 20));   // 32*128 f32
    float*  logdet = tvec + GK * GDIM;           // 32 f32 (+pad)
    float*  parts  = logdet + 64;                // 32 * gx f32

    const int gx = (N + 127) / 128;

    hipLaunchKernelGGL(stageA_kernel, dim3(GK), dim3(128), 0, stream,
                       cov, mu, Mh, Ml, tvec, logdet);
    hipLaunchKernelGGL(maha_kernel, dim3(gx, GK), dim3(128), 0, stream,
                       X, Mh, Ml, pvec, tvec, logdet, parts, N);
    hipLaunchKernelGGL(finalize_kernel, dim3(1), dim3(256), 0, stream,
                       parts, GK * gx, (float*)d_out);
}
